// DeformableCrossAttention_67568425500978
// MI455X (gfx1250) — matrix operand order repaired and checked by reading, not compiled, not run on hardware
//
#include <hip/hip_runtime.h>
#include <hip/hip_bf16.h>
#include <math.h>

typedef __attribute__((ext_vector_type(16))) _Float16 v16h;
typedef __attribute__((ext_vector_type(8)))  _Float16 v8h;
typedef __attribute__((ext_vector_type(8)))  float    v8f;
typedef __attribute__((ext_vector_type(4)))  float    v4f;

// Problem constants (match reference)
constexpr int Bc = 2, Qc = 4096, Cc = 256, Nn = 6, Hh = 64, Ww = 96;
constexpr int HEADS = 8, POINTS = 4, HD = Cc / HEADS;    // 32
constexpr int HWc  = Hh * Ww;                            // 6144
constexpr int NOFF = Nn * HEADS * POINTS * 2;            // 384
constexpr int NATT = Nn * HEADS * POINTS;                // 192
constexpr int NCAT = NOFF + NATT;                        // 576

// --- CDNA5 async global->LDS helpers --------------------------------------
__device__ __forceinline__ void async_ld_b128(unsigned lds_off, const void* gaddr) {
  // GLOBAL_LOAD_ASYNC_TO_LDS_B128 (GV mode): per-lane 16B memory -> LDS copy,
  // tracked by ASYNCcnt.
  asm volatile("global_load_async_to_lds_b128 %0, %1, off"
               :: "v"(lds_off), "v"(gaddr) : "memory");
}
__device__ __forceinline__ void wait_async0() {
#if __has_builtin(__builtin_amdgcn_s_wait_asynccnt)
  __builtin_amdgcn_s_wait_asynccnt(0);
#else
  asm volatile("s_wait_asynccnt 0x0" ::: "memory");
#endif
}

// ---------------------------------------------------------------------------
// K0: pack fp32 weights to fp16 in COLUMN-MAJOR [n][k] layout so GEMM B-tiles
// are contiguous 64B rows (async-copyable straight into the LDS layout the
// WMMA B-fragment reads want). Also concatenates Woff|Wattn and their biases.
// ---------------------------------------------------------------------------
__global__ void pack_weights(const float* __restrict__ Woff, const float* __restrict__ boff,
                             const float* __restrict__ Wattn, const float* __restrict__ battn,
                             const float* __restrict__ Wval, const float* __restrict__ Wout,
                             _Float16* __restrict__ Wcat, float* __restrict__ bcat,
                             _Float16* __restrict__ Wv, _Float16* __restrict__ Wo) {
  int i = blockIdx.x * blockDim.x + threadIdx.x;
  const int nWcat = Cc * NCAT;
  if (i < nWcat) {                       // Wcat[n][k], n in [0,576), k in [0,256)
    int n = i / Cc, k = i % Cc;
    float v = (n < NOFF) ? Woff[k * NOFF + n] : Wattn[k * NATT + (n - NOFF)];
    Wcat[i] = (_Float16)v;
    return;
  }
  int j2 = i - nWcat;
  if (j2 < Cc * Cc) {                    // Wv[d][c] = Wval[c][d]
    int d = j2 / Cc, c = j2 % Cc;
    Wv[j2] = (_Float16)Wval[c * Cc + d];
    return;
  }
  int j3 = j2 - Cc * Cc;
  if (j3 < Cc * Cc) {                    // Wo[d][c] = Wout[c][d]
    int d = j3 / Cc, c = j3 % Cc;
    Wo[j3] = (_Float16)Wout[c * Cc + d];
    return;
  }
  int j4 = j3 - Cc * Cc;
  if (j4 < NCAT) bcat[j4] = (j4 < NOFF) ? boff[j4] : battn[j4 - NOFF];
}

// ---------------------------------------------------------------------------
// K1: voxel -> world -> camera -> f-theta fisheye reference points + fov mask
// ---------------------------------------------------------------------------
__global__ void ref_points(const float* __restrict__ qc, const float* __restrict__ intr,
                           const float* __restrict__ invE,
                           float* __restrict__ refBuf, float* __restrict__ fovBuf) {
  int i = blockIdx.x * blockDim.x + threadIdx.x;
  if (i >= Bc * Nn * Qc) return;
  int q = i % Qc; int bn = i / Qc; int n = bn % Nn; int b = bn / Nn;
  const float* c3 = qc + ((size_t)b * Qc + q) * 3;
  float rx = c3[0] * 7.2f - 3.6f;
  float ry = c3[1] * 6.0f - 3.0f;
  float rz = c3[2] * 3.2f - 0.7f;
  const float* E = invE + (size_t)(b * Nn + n) * 16;
  float cxp = E[0] * rx + E[1] * ry + E[2]  * rz + E[3];
  float cyp = E[4] * rx + E[5] * ry + E[6]  * rz + E[7];
  float czp = E[8] * rx + E[9] * ry + E[10] * rz + E[11];
  float r3d   = sqrtf(cxp * cxp + cyp * cyp + 1e-8f);
  float theta = atan2f(r3d, czp);
  float phi   = atan2f(cyp, cxp);
  const float* Km = intr + (size_t)(b * Nn + n) * 9;
  float fx = Km[0], cx0 = Km[2], cy0 = Km[5];
  float r_img = fx * theta;
  float u = r_img * cosf(phi) + cx0;
  float v = r_img * sinf(phi) + cy0;
  refBuf[(size_t)i * 2 + 0] = 2.0f * u / 1279.0f - 1.0f;
  refBuf[(size_t)i * 2 + 1] = 2.0f * v / 1079.0f - 1.0f;
  fovBuf[i] = (czp > 0.0f && theta < 1.372f) ? 1.0f : 0.0f;
}

// ---------------------------------------------------------------------------
// Shared W-tile stage: async-copy a 64(n) x 32(k) fp16 tile (Wp column-major
// [Ncols][256]) into LDS [n][k]. 256 threads x one async b128 each = 4KB.
// ---------------------------------------------------------------------------
__device__ __forceinline__ void stage_w_async(_Float16* ldsW, const _Float16* __restrict__ Wp,
                                              int n0, int k0) {
  const int nn = threadIdx.x >> 2;         // 0..63
  const int kc = (threadIdx.x & 3) * 8;    // 0,8,16,24
  unsigned lds_off = (unsigned)(size_t)(&ldsW[nn * 32 + kc]);
  const _Float16* gsrc = Wp + (size_t)(n0 + nn) * Cc + k0 + kc;
  async_ld_b128(lds_off, (const void*)gsrc);
  wait_async0();
}

// ---------------------------------------------------------------------------
// WMMA GEMM, A row-major fp32 [M x 256], Wp fp16 column-major [Ncols x 256],
// out fp32. Wave: 16x64 tile (4x v_wmma_f32_16x16x32_f16 per K-step).
// Block: 8 waves stacked in M (128-row strip) share the async-staged W tile.
// ---------------------------------------------------------------------------
__global__ void __launch_bounds__(256) wmma_gemm_rm(const float* __restrict__ A,
                                                    const _Float16* __restrict__ Wp,
                                                    const float* __restrict__ bias,
                                                    float* __restrict__ Out, int Ncols) {
  __shared__ alignas(64) _Float16 ldsW[64 * 32];   // [n within 64][k within 32]
  const int lane = threadIdx.x & 31;
  const int wave = threadIdx.x >> 5;
  const int n0   = blockIdx.x * 64;
  const int m0   = blockIdx.y * 128 + wave * 16;
  const int row  = m0 + (lane & 15);
  const int coll = lane & 15;
  const int hi   = lane >> 4;

  v8f acc[4] = {};
  for (int k0 = 0; k0 < Cc; k0 += 32) {
    stage_w_async(ldsW, Wp, n0, k0);
    __syncthreads();

    // A fragment: contiguous k order, element s of lane half hi is k = 16*hi + s (same as B)
    const float* ap = A + (size_t)row * Cc + k0 + (hi << 4);
    v4f a0 = *(const v4f*)(ap);
    v4f a1 = *(const v4f*)(ap + 4);
    v4f a2 = *(const v4f*)(ap + 8);
    v4f a3 = *(const v4f*)(ap + 12);
    v16h a;
#pragma unroll
    for (int j = 0; j < 4; j++) {
      a[j]      = (_Float16)a0[j];
      a[4 + j]  = (_Float16)a1[j];
      a[8 + j]  = (_Float16)a2[j];
      a[12 + j] = (_Float16)a3[j];
    }
    const int koff = hi << 4;
#pragma unroll
    for (int c = 0; c < 4; c++) {
      v16h bf = *(const v16h*)(&ldsW[(c * 16 + coll) * 32 + koff]);
      acc[c] = __builtin_amdgcn_wmma_f32_16x16x32_f16(
          false, a, false, bf, (short)0, acc[c], false, false);
    }
    __syncthreads();
  }

  const int mb = hi * 8;
#pragma unroll
  for (int c = 0; c < 4; c++) {
    int col = n0 + c * 16 + coll;
    float bv = bias[col];
#pragma unroll
    for (int i = 0; i < 8; i++)
      Out[(size_t)(m0 + mb + i) * Ncols + col] = acc[c][i] + bv;
  }
}

// ---------------------------------------------------------------------------
// Value projection: einsum('bnchw,cd->bnhwd'). A is image_feats with K
// (channel) stride HW (transposed access); output val is fp16 pixel-major
// [B,N,HW,256] so the sampler gathers contiguous 64B per (head, tap).
// ---------------------------------------------------------------------------
__global__ void __launch_bounds__(256) wmma_gemm_val(const float* __restrict__ feats,
                                                     const _Float16* __restrict__ Wv,
                                                     const float* __restrict__ bval,
                                                     _Float16* __restrict__ val) {
  __shared__ alignas(64) _Float16 ldsW[64 * 32];
  const int lane = threadIdx.x & 31;
  const int wave = threadIdx.x >> 5;
  const int n0   = blockIdx.x * 64;
  const int m0   = blockIdx.y * 128 + wave * 16;     // global pixel-row base
  const int gRow = m0 + (lane & 15);
  const int bn   = gRow / HWc;                       // 128 | 6144, strip stays in one (b,n)
  const int hw   = gRow % HWc;
  const float* abase = feats + (size_t)bn * Cc * HWc + hw;
  const int coll = lane & 15;
  const int hi   = lane >> 4;

  v8f acc[4] = {};
  for (int k0 = 0; k0 < Cc; k0 += 32) {
    stage_w_async(ldsW, Wv, n0, k0);
    __syncthreads();

    const int kb = k0 + (hi << 4);
    v16h a;
#pragma unroll
    for (int j = 0; j < 8; j++) a[j]     = (_Float16)abase[(size_t)(kb + j) * HWc];
#pragma unroll
    for (int j = 0; j < 8; j++) a[8 + j] = (_Float16)abase[(size_t)(kb + 8 + j) * HWc];

    const int koff = hi << 4;
#pragma unroll
    for (int c = 0; c < 4; c++) {
      v16h bf = *(const v16h*)(&ldsW[(c * 16 + coll) * 32 + koff]);
      acc[c] = __builtin_amdgcn_wmma_f32_16x16x32_f16(
          false, a, false, bf, (short)0, acc[c], false, false);
    }
    __syncthreads();
  }

  const int mb = hi * 8;
#pragma unroll
  for (int c = 0; c < 4; c++) {
    int col = n0 + c * 16 + coll;
    float bv = bval[col];
#pragma unroll
    for (int i = 0; i < 8; i++)
      val[(size_t)(m0 + mb + i) * Cc + col] = (_Float16)(acc[c][i] + bv);
  }
}

// ---------------------------------------------------------------------------
// K4: fused tanh-offsets + fov-masked softmax + bilinear sampling + reduce
// over points & cameras. One thread per (b,q,head) owns 32 fp32 channels.
// ---------------------------------------------------------------------------
__global__ void __launch_bounds__(256) sample_agg(const float* __restrict__ raw,
                                                  const float* __restrict__ refBuf,
                                                  const float* __restrict__ fovBuf,
                                                  const _Float16* __restrict__ val,
                                                  float* __restrict__ agg) {
  int tid = blockIdx.x * blockDim.x + threadIdx.x;
  if (tid >= Bc * Qc * HEADS) return;
  const int head = tid % HEADS;
  const int bq   = tid / HEADS;
  const int q    = bq % Qc;
  const int b    = bq / Qc;
  const float* rw = raw + (size_t)bq * NCAT;

  float acc[HD];
#pragma unroll
  for (int ch = 0; ch < HD; ch++) acc[ch] = 0.0f;

  for (int n = 0; n < Nn; n++) {
    const int bnq = (b * Nn + n) * Qc + q;
    const float fov  = fovBuf[bnq];
    const float refx = refBuf[(size_t)bnq * 2 + 0];
    const float refy = refBuf[(size_t)bnq * 2 + 1];

    // fov-masked softmax over 4 points (fov==0 -> uniform 0.25, as reference)
    const int abase = NOFF + (n * HEADS + head) * POINTS;
    float lg[POINTS], wp[POINTS];
    float mx = -1e30f;
#pragma unroll
    for (int p = 0; p < POINTS; p++) { lg[p] = rw[abase + p] * fov; mx = fmaxf(mx, lg[p]); }
    float s = 0.0f;
#pragma unroll
    for (int p = 0; p < POINTS; p++) { wp[p] = __expf(lg[p] - mx); s += wp[p]; }
    const float inv = 1.0f / s;

    const int obase = (n * HEADS + head) * POINTS * 2;
    const _Float16* vb = val + (size_t)(b * Nn + n) * HWc * Cc + head * HD;

#pragma unroll
    for (int p = 0; p < POINTS; p++) {
      const float w_p = wp[p] * inv;
      const float ox = tanhf(rw[obase + p * 2 + 0]) * 0.5f;
      const float oy = tanhf(rw[obase + p * 2 + 1]) * 0.5f;
      const float gx = (refx + ox + 1.0f) * (Ww * 0.5f) - 0.5f;
      const float gy = (refy + oy + 1.0f) * (Hh * 0.5f) - 0.5f;
      const float x0f = floorf(gx), y0f = floorf(gy);
      const float wx = gx - x0f, wy = gy - y0f;
      const int x0 = (int)x0f, y0 = (int)y0f;
      const float cw[4] = { (1.0f - wx) * (1.0f - wy) * w_p,
                            wx * (1.0f - wy) * w_p,
                            (1.0f - wx) * wy * w_p,
                            wx * wy * w_p };
      const int cx[4] = { x0, x0 + 1, x0, x0 + 1 };
      const int cy[4] = { y0, y0, y0 + 1, y0 + 1 };
#pragma unroll
      for (int t = 0; t < 4; t++) {
        const int xi = cx[t], yi = cy[t];
        if (xi < 0 || xi >= Ww || yi < 0 || yi >= Hh) continue;  // zero padding
        const float wgt = cw[t];
        const _Float16* p0 = vb + (size_t)(yi * Ww + xi) * Cc;
        v8h c0 = *(const v8h*)(p0);
        v8h c1 = *(const v8h*)(p0 + 8);
        v8h c2 = *(const v8h*)(p0 + 16);
        v8h c3 = *(const v8h*)(p0 + 24);
#pragma unroll
        for (int j = 0; j < 8; j++) {
          acc[j]      = fmaf(wgt, (float)c0[j], acc[j]);
          acc[8 + j]  = fmaf(wgt, (float)c1[j], acc[8 + j]);
          acc[16 + j] = fmaf(wgt, (float)c2[j], acc[16 + j]);
          acc[24 + j] = fmaf(wgt, (float)c3[j], acc[24 + j]);
        }
      }
    }
  }

  float* op = agg + (size_t)bq * Cc + head * HD;
#pragma unroll
  for (int j = 0; j < HD; j += 4) {
    v4f t = { acc[j], acc[j + 1], acc[j + 2], acc[j + 3] };
    *(v4f*)(op + j) = t;
  }
}

// ---------------------------------------------------------------------------
extern "C" void kernel_launch(void* const* d_in, const int* in_sizes, int n_in,
                              void* d_out, int out_size, void* d_ws, size_t ws_size,
                              hipStream_t stream) {
  const float* query   = (const float*)d_in[0];
  const float* qcoords = (const float*)d_in[1];
  const float* feats   = (const float*)d_in[2];
  const float* intr    = (const float*)d_in[3];
  const float* invE    = (const float*)d_in[4];
  const float* Woff    = (const float*)d_in[5];
  const float* boff    = (const float*)d_in[6];
  const float* Wattn   = (const float*)d_in[7];
  const float* battn   = (const float*)d_in[8];
  const float* Wval    = (const float*)d_in[9];
  const float* bval    = (const float*)d_in[10];
  const float* Wout    = (const float*)d_in[11];
  const float* bout    = (const float*)d_in[12];

  char* base = (char*)d_ws;
  size_t off = 0;
  auto alloc = [&](size_t bytes) -> char* {
    char* p = base + off;
    off += (bytes + 255) & ~(size_t)255;
    return p;
  };
  _Float16* Wcat  = (_Float16*)alloc((size_t)Cc * NCAT * sizeof(_Float16));
  float*    bcat  = (float*)   alloc((size_t)NCAT * sizeof(float));
  _Float16* Wv    = (_Float16*)alloc((size_t)Cc * Cc * sizeof(_Float16));
  _Float16* Wo    = (_Float16*)alloc((size_t)Cc * Cc * sizeof(_Float16));
  float*    refB  = (float*)   alloc((size_t)Bc * Nn * Qc * 2 * sizeof(float));
  float*    fovB  = (float*)   alloc((size_t)Bc * Nn * Qc * sizeof(float));
  float*    raw   = (float*)   alloc((size_t)Bc * Qc * NCAT * sizeof(float));
  _Float16* val   = (_Float16*)alloc((size_t)Bc * Nn * HWc * Cc * sizeof(_Float16));
  float*    agg   = (float*)   alloc((size_t)Bc * Qc * Cc * sizeof(float));

  // K0: weight packing to fp16 (column-major for async B-tile staging)
  const int totPack = Cc * NCAT + 2 * Cc * Cc + NCAT;
  pack_weights<<<(totPack + 255) / 256, 256, 0, stream>>>(
      Woff, boff, Wattn, battn, Wval, Wout, Wcat, bcat, Wv, Wo);

  // K1: reference points + fov mask
  ref_points<<<(Bc * Nn * Qc + 255) / 256, 256, 0, stream>>>(
      qcoords, intr, invE, refB, fovB);

  // K2: offsets+attn projection  [8192x256] x [256x576]
  wmma_gemm_rm<<<dim3(NCAT / 64, (Bc * Qc) / 128), 256, 0, stream>>>(
      query, Wcat, bcat, raw, NCAT);

  // K3: value projection  [73728x256] x [256x256] -> fp16 val
  wmma_gemm_val<<<dim3(Cc / 64, (Bc * Nn * HWc) / 128), 256, 0, stream>>>(
      feats, Wv, bval, val);

  // K4: fused softmax + bilinear sampling + point/camera reduction
  sample_agg<<<(Bc * Qc * HEADS + 255) / 256, 256, 0, stream>>>(
      raw, refB, fovB, val, agg);

  // K5: output projection  [8192x256] x [256x256] -> d_out
  wmma_gemm_rm<<<dim3(Cc / 64, (Bc * Qc) / 128), 256, 0, stream>>>(
      agg, Wo, bout, (float*)d_out, Cc);
}
